// MultiHeadAttention_88072599371993
// MI455X (gfx1250) — compile-verified, matrix operand order checked by reading, not run on hardware
//
#include <hip/hip_runtime.h>
#include <hip/hip_bf16.h>

typedef __attribute__((ext_vector_type(16))) _Float16 v16h;
typedef __attribute__((ext_vector_type(8)))  _Float16 v8h;
typedef __attribute__((ext_vector_type(4)))  _Float16 v4h;
typedef __attribute__((ext_vector_type(8)))  float    v8f;

#define B_  4
#define S_  2048
#define D_  1024
#define H_  16
#define DK_ 64

// ---------------------------------------------------------------------------
// gfx1250 async copy to LDS (ASYNCcnt path). Guarded: falls back to a
// synchronous b128 copy if the builtin is not available in this toolchain.
// Builtin signature (from hipcc diagnostic): param0 = int4 addrspace(1)*,
// so it is (global_src, lds_dst, imm_offset, imm_cpol) with int4 pointees.
// ---------------------------------------------------------------------------
#if defined(__gfx1250__) && __has_builtin(__builtin_amdgcn_global_load_async_to_lds_b128)
#define ASYNC_LDS 1
#else
#define ASYNC_LDS 0
#endif

#if ASYNC_LDS
typedef int v4i_ __attribute__((vector_size(4 * sizeof(int))));
#endif

__device__ __forceinline__ void cp_async_b128(const _Float16* gsrc,
                                              _Float16* ldst) {
#if ASYNC_LDS
  __builtin_amdgcn_global_load_async_to_lds_b128(
      (__attribute__((address_space(1))) v4i_*)gsrc,
      (__attribute__((address_space(3))) v4i_*)ldst, 0, 0);
#else
  *(v8h*)ldst = *(const v8h*)gsrc;
#endif
}

__device__ __forceinline__ void async_wait_all() {
#if ASYNC_LDS && __has_builtin(__builtin_amdgcn_s_wait_asynccnt)
  __builtin_amdgcn_s_wait_asynccnt(0);
#endif
}

__device__ __forceinline__ void prefetch_g(const void* p) {
  __builtin_prefetch(p, 0, 3);  // lowers to global_prefetch_b8 on gfx1250
}

// ---------------------------------------------------------------------------
// WMMA helpers (CDNA5: V_WMMA_F32_16X16X32_F16, wave32)
// ---------------------------------------------------------------------------
__device__ __forceinline__ v8f wmma_f16(v16h a, v16h b, v8f c) {
  return __builtin_amdgcn_wmma_f32_16x16x32_f16(false, a, false, b, (short)0, c,
                                                false, false);
}

// Load a 16x32 f16 A/B fragment from an LDS tile stored row-major [16][stride].
// ISA layout: lane L -> row L%16; halves 0..7 = k (L/16)*8+{0..7},
//             halves 8..15 = k 16+(L/16)*8+{0..7}.
__device__ __forceinline__ v16h ldfrag(const _Float16* base, int stride, int lane) {
  const int row = lane & 15;
  const int hw  = lane >> 4;
  const _Float16* p = base + row * stride + hw * 8;
  v8h a = *(const v8h*)(p);
  v8h b = *(const v8h*)(p + 16);
  v16h r;
#pragma unroll
  for (int i = 0; i < 8; ++i) { r[i] = a[i]; r[i + 8] = b[i]; }
  return r;
}

// ---------------------------------------------------------------------------
// f32 -> f16 conversion (memory-bound, vectorized x4)
// ---------------------------------------------------------------------------
__global__ __launch_bounds__(256) void cvt_f32_f16(const float* __restrict__ in,
                                                   _Float16* __restrict__ out,
                                                   int n) {
  int i = (blockIdx.x * 256 + threadIdx.x) * 4;
  if (i + 3 < n) {
    float4 v = *(const float4*)(in + i);
    v4h o;
    o[0] = (_Float16)v.x; o[1] = (_Float16)v.y;
    o[2] = (_Float16)v.z; o[3] = (_Float16)v.w;
    *(v4h*)(out + i) = o;
  }
}

// ---------------------------------------------------------------------------
// Projection GEMM: C[M,N] = A[M,K] * W[N,K]^T + bias
//   MODE 0: f16 head-split   out[((b*H+h)*S+s)*DK+dk]      (Q,K proj)
//   MODE 1: f16 head-transp  out[((b*H+h)*DK+dk)*S+s]      (V proj)
//   MODE 2: f32 + residual   out[m*D+n]                    (out proj)
// Block tile 128x64, BK=32; 8 waves (4x2), wave tile 32x32 -> 4 WMMAs/K-step.
// ---------------------------------------------------------------------------
template <int MODE>
__global__ __launch_bounds__(256) void proj_gemm(
    const _Float16* __restrict__ A, const _Float16* __restrict__ W,
    const float* __restrict__ bias, const float* __restrict__ residual,
    _Float16* __restrict__ out_h, float* __restrict__ out_f) {
  __shared__ alignas(16) _Float16 As[128 * 40];
  __shared__ alignas(16) _Float16 Bs[64 * 40];

  const int tid  = threadIdx.x;
  const int lane = tid & 31;
  const int w    = tid >> 5;   // wave 0..7
  const int wm   = w >> 1;     // 0..3 (32-row slice)
  const int wn   = w & 1;      // 0..1 (32-col slice)
  const int n0   = blockIdx.x * 64;
  const int m0   = blockIdx.y * 128;

  v8f c[2][2] = {{{}, {}}, {{}, {}}};

  for (int kt = 0; kt < D_; kt += 32) {
#pragma unroll
    for (int i = 0; i < 2; ++i) {  // A tile: 128 rows x 4 chunks
      const int idx = tid + i * 256;
      const int r = idx >> 2, ch = (idx & 3) * 8;
      cp_async_b128(A + (size_t)(m0 + r) * D_ + kt + ch, &As[r * 40 + ch]);
    }
    {  // B tile: 64 rows x 4 chunks
      const int r = tid >> 2, ch = (tid & 3) * 8;
      cp_async_b128(W + (size_t)(n0 + r) * D_ + kt + ch, &Bs[r * 40 + ch]);
    }
    async_wait_all();
    __syncthreads();

    v16h a0 = ldfrag(&As[(wm * 32) * 40], 40, lane);
    v16h a1 = ldfrag(&As[(wm * 32 + 16) * 40], 40, lane);
    v16h b0 = ldfrag(&Bs[(wn * 32) * 40], 40, lane);
    v16h b1 = ldfrag(&Bs[(wn * 32 + 16) * 40], 40, lane);
    c[0][0] = wmma_f16(a0, b0, c[0][0]);
    c[0][1] = wmma_f16(a0, b1, c[0][1]);
    c[1][0] = wmma_f16(a1, b0, c[1][0]);
    c[1][1] = wmma_f16(a1, b1, c[1][1]);
    __syncthreads();
  }

  const int hlf = lane >> 4;
  const int col = lane & 15;
#pragma unroll
  for (int am = 0; am < 2; ++am) {
#pragma unroll
    for (int bn = 0; bn < 2; ++bn) {
#pragma unroll
      for (int v = 0; v < 8; ++v) {
        const int m = m0 + wm * 32 + am * 16 + v + hlf * 8;
        const int n = n0 + wn * 32 + bn * 16 + col;
        float val = c[am][bn][v] + bias[n];
        if (MODE == 2) {
          const size_t idx = (size_t)m * D_ + n;
          out_f[idx] = val + residual[idx];
        } else {
          const int bb = m / S_, s = m - bb * S_;
          const int h = n >> 6, dk = n & 63;  // DK_ = 64
          if (MODE == 0)
            out_h[(((size_t)bb * H_ + h) * S_ + s) * DK_ + dk] = (_Float16)val;
          else
            out_h[(((size_t)bb * H_ + h) * DK_ + dk) * S_ + s] = (_Float16)val;
        }
      }
    }
  }
}

// ---------------------------------------------------------------------------
// Flash attention: one block = 128 Q rows of one (b,h); 8 waves x 16 Q rows.
// Streams 32-key tiles; scores never touch HBM (B*H*S*S f32 scores would be 1 GiB).
// qh: [B,H,S,DK] f16, kh: [B,H,S,DK] f16, vt: [B,H,DK,S] f16 (pre-transposed),
// attn out: [B,S,D] f16 (heads re-merged).
// ---------------------------------------------------------------------------
__global__ __launch_bounds__(256) void flash_attn(
    const _Float16* __restrict__ qh, const _Float16* __restrict__ kh,
    const _Float16* __restrict__ vt, const int* __restrict__ mask,
    _Float16* __restrict__ attn) {
  __shared__ alignas(16) _Float16 Qs[128 * 72];   // [qrow][dk]
  __shared__ alignas(16) _Float16 Ks[32 * 72];    // [key ][dk]
  __shared__ alignas(16) _Float16 Vts[64 * 40];   // [dk  ][key]
  __shared__ alignas(16) _Float16 Ps[8][16 * 40]; // per-wave probs [q][key]

  const int tid  = threadIdx.x;
  const int lane = tid & 31;
  const int w    = tid >> 5;
  const int qb   = blockIdx.x * 128;
  const int hh   = blockIdx.y;
  const int bb   = blockIdx.z;

  const size_t head = (size_t)bb * H_ + hh;
  const _Float16* qg = qh + head * S_ * DK_;
  const _Float16* kg = kh + head * S_ * DK_;
  const _Float16* vg = vt + head * DK_ * S_;

  // Stage Q block (128x64 halves) once (async; waited with first tile).
#pragma unroll
  for (int i = 0; i < 4; ++i) {
    const int idx = tid + i * 256;          // 0..1023 chunks of 8 halves
    const int r = idx >> 3, ch = (idx & 7) * 8;
    cp_async_b128(qg + (size_t)(qb + r) * DK_ + ch, &Qs[r * 72 + ch]);
  }

  v8f o0 = {}, o1 = {}, o2 = {}, o3 = {};
  float mrow[8], lrow[8];
#pragma unroll
  for (int v = 0; v < 8; ++v) { mrow[v] = -1e30f; lrow[v] = 0.f; }

  const int hlf = lane >> 4;
  const int col = lane & 15;

  const int kr = tid >> 3, kch = (tid & 7) * 8;   // K-tile staging slot
  const int vr = tid >> 2, vch = (tid & 3) * 8;   // V-tile staging slot

  for (int kt = 0; kt < S_; kt += 32) {
    cp_async_b128(kg + (size_t)(kt + kr) * DK_ + kch, &Ks[kr * 72 + kch]);
    cp_async_b128(vg + (size_t)vr * S_ + kt + vch, &Vts[vr * 40 + vch]);
    if (kt + 32 < S_) {  // prefetch next key tile into L2/WGP$
      prefetch_g(kg + (size_t)(kt + 32 + kr) * DK_ + kch);
      prefetch_g(vg + (size_t)vr * S_ + kt + 32 + vch);
    }
    async_wait_all();
    __syncthreads();

    // scores(16q x 32k) = Q(16x64) * K^T(64x32): 4 WMMAs
    v8f s0 = {}, s1 = {};
    v16h aq0 = ldfrag(&Qs[w * 16 * 72], 72, lane);
    v16h aq1 = ldfrag(&Qs[w * 16 * 72 + 32], 72, lane);
    s0 = wmma_f16(aq0, ldfrag(&Ks[0], 72, lane), s0);
    s0 = wmma_f16(aq1, ldfrag(&Ks[32], 72, lane), s0);
    s1 = wmma_f16(aq0, ldfrag(&Ks[16 * 72], 72, lane), s1);
    s1 = wmma_f16(aq1, ldfrag(&Ks[16 * 72 + 32], 72, lane), s1);

    // online softmax (per row; reduce across the 16 lanes holding one row)
#pragma unroll
    for (int v = 0; v < 8; ++v) {
      const int qrow = qb + w * 16 + v + hlf * 8;
      const int key0 = kt + col;
      const int mv0 = mask[(size_t)qrow * S_ + key0];
      const int mv1 = mask[(size_t)qrow * S_ + key0 + 16];
      float x0 = mv0 ? s0[v] * 0.125f : -1e9f;  // 1/sqrt(64)
      float x1 = mv1 ? s1[v] * 0.125f : -1e9f;
      float mx = fmaxf(x0, x1);
#pragma unroll
      for (int off = 8; off >= 1; off >>= 1)
        mx = fmaxf(mx, __shfl_xor(mx, off, 32));
      const float mn = fmaxf(mrow[v], mx);
      const float al = __expf(mrow[v] - mn);
      const float p0 = mv0 ? __expf(x0 - mn) : 0.f;
      const float p1 = mv1 ? __expf(x1 - mn) : 0.f;
      float ps = p0 + p1;
#pragma unroll
      for (int off = 8; off >= 1; off >>= 1) ps += __shfl_xor(ps, off, 32);
      lrow[v] = lrow[v] * al + ps;
      mrow[v] = mn;
      o0[v] *= al; o1[v] *= al; o2[v] *= al; o3[v] *= al;
      const int pr = v + hlf * 8;  // C-layout -> LDS [q][key] for A-fragment
      Ps[w][pr * 40 + col]      = (_Float16)p0;
      Ps[w][pr * 40 + 16 + col] = (_Float16)p1;
    }

    // O(16q x 64dk) += P(16x32) * V(32x64): 4 WMMAs.
    // Ps is private to this wave: wave-synchronous DS ordering, no barrier.
    v16h ap = ldfrag(&Ps[w][0], 40, lane);
    o0 = wmma_f16(ap, ldfrag(&Vts[0],       40, lane), o0);
    o1 = wmma_f16(ap, ldfrag(&Vts[16 * 40], 40, lane), o1);
    o2 = wmma_f16(ap, ldfrag(&Vts[32 * 40], 40, lane), o2);
    o3 = wmma_f16(ap, ldfrag(&Vts[48 * 40], 40, lane), o3);
    __syncthreads();
  }

  // Normalize and write merged-head layout [B,S,D].
#pragma unroll
  for (int v = 0; v < 8; ++v) {
    const int q = qb + w * 16 + v + hlf * 8;
    const float inv = 1.f / (lrow[v] + 1e-9f);
    const size_t base = ((size_t)bb * S_ + q) * D_ + hh * DK_;
    attn[base +  0 + col] = (_Float16)(o0[v] * inv);
    attn[base + 16 + col] = (_Float16)(o1[v] * inv);
    attn[base + 32 + col] = (_Float16)(o2[v] * inv);
    attn[base + 48 + col] = (_Float16)(o3[v] * inv);
  }
}

// ---------------------------------------------------------------------------
// LayerNorm over D=1024 per row; 256 threads x 4 elems.
// ---------------------------------------------------------------------------
__global__ __launch_bounds__(256) void layernorm_kernel(
    const float* __restrict__ x, const float* __restrict__ g,
    const float* __restrict__ b, float* __restrict__ out) {
  __shared__ float red[16];
  const int row = blockIdx.x;
  const int tid = threadIdx.x;
  const float4 v = *(const float4*)(x + (size_t)row * D_ + tid * 4);
  float s  = v.x + v.y + v.z + v.w;
  float q2 = v.x * v.x + v.y * v.y + v.z * v.z + v.w * v.w;
#pragma unroll
  for (int off = 16; off >= 1; off >>= 1) {
    s  += __shfl_xor(s, off, 32);
    q2 += __shfl_xor(q2, off, 32);
  }
  const int w = tid >> 5, lane = tid & 31;
  if (lane == 0) { red[w] = s; red[8 + w] = q2; }
  __syncthreads();
  float ts = 0.f, tq = 0.f;
#pragma unroll
  for (int i = 0; i < 8; ++i) { ts += red[i]; tq += red[8 + i]; }
  const float mu   = ts * (1.f / D_);
  const float var  = tq * (1.f / D_) - mu * mu;
  const float rstd = rsqrtf(var + 1e-5f);
  const float4 gg = *(const float4*)(g + tid * 4);
  const float4 bb = *(const float4*)(b + tid * 4);
  float4 o;
  o.x = (v.x - mu) * rstd * gg.x + bb.x;
  o.y = (v.y - mu) * rstd * gg.y + bb.y;
  o.z = (v.z - mu) * rstd * gg.z + bb.z;
  o.w = (v.w - mu) * rstd * gg.w + bb.w;
  *(float4*)(out + (size_t)row * D_ + tid * 4) = o;
}

// ---------------------------------------------------------------------------
// Launch
// ---------------------------------------------------------------------------
extern "C" void kernel_launch(void* const* d_in, const int* in_sizes, int n_in,
                              void* d_out, int out_size, void* d_ws,
                              size_t ws_size, hipStream_t stream) {
  const float* q    = (const float*)d_in[0];
  const float* k    = (const float*)d_in[1];
  const float* v    = (const float*)d_in[2];
  const int*   mask = (const int*)d_in[3];
  const float* Wq = (const float*)d_in[4];
  const float* bq = (const float*)d_in[5];
  const float* Wk = (const float*)d_in[6];
  const float* bk = (const float*)d_in[7];
  const float* Wv = (const float*)d_in[8];
  const float* bv = (const float*)d_in[9];
  const float* Wo = (const float*)d_in[10];
  const float* bo = (const float*)d_in[11];
  const float* lng = (const float*)d_in[12];
  const float* lnb = (const float*)d_in[13];
  float* out = (float*)d_out;

  // Workspace layout (all 16B-aligned, total ~152 MB)
  char* ws = (char*)d_ws;
  const size_t ACT = (size_t)B_ * S_ * D_ * sizeof(_Float16);  // 16 MB
  const size_t WSZ = (size_t)D_ * D_ * sizeof(_Float16);       //  2 MB
  _Float16* qf   = (_Float16*)(ws);
  _Float16* kf   = (_Float16*)(ws + ACT);
  _Float16* vf   = (_Float16*)(ws + 2 * ACT);
  _Float16* wqh  = (_Float16*)(ws + 3 * ACT);
  _Float16* wkh  = (_Float16*)(ws + 3 * ACT + WSZ);
  _Float16* wvh  = (_Float16*)(ws + 3 * ACT + 2 * WSZ);
  _Float16* woh  = (_Float16*)(ws + 3 * ACT + 3 * WSZ);
  _Float16* qhd  = (_Float16*)(ws + 3 * ACT + 4 * WSZ);
  _Float16* khd  = (_Float16*)(ws + 4 * ACT + 4 * WSZ);
  _Float16* vtd  = (_Float16*)(ws + 5 * ACT + 4 * WSZ);
  _Float16* atth = (_Float16*)(ws + 6 * ACT + 4 * WSZ);
  float*    proj = (float*)   (ws + 7 * ACT + 4 * WSZ);

  const int NT = 256;
  const int nact = B_ * S_ * D_;
  const int nw   = D_ * D_;
  cvt_f32_f16<<<nact / (NT * 4), NT, 0, stream>>>(q,  qf,  nact);
  cvt_f32_f16<<<nact / (NT * 4), NT, 0, stream>>>(k,  kf,  nact);
  cvt_f32_f16<<<nact / (NT * 4), NT, 0, stream>>>(v,  vf,  nact);
  cvt_f32_f16<<<nw   / (NT * 4), NT, 0, stream>>>(Wq, wqh, nw);
  cvt_f32_f16<<<nw   / (NT * 4), NT, 0, stream>>>(Wk, wkh, nw);
  cvt_f32_f16<<<nw   / (NT * 4), NT, 0, stream>>>(Wv, wvh, nw);
  cvt_f32_f16<<<nw   / (NT * 4), NT, 0, stream>>>(Wo, woh, nw);

  dim3 gg(D_ / 64, (B_ * S_) / 128);  // (16,64)
  proj_gemm<0><<<gg, NT, 0, stream>>>(qf, wqh, bq, nullptr, qhd, nullptr);
  proj_gemm<0><<<gg, NT, 0, stream>>>(kf, wkh, bk, nullptr, khd, nullptr);
  proj_gemm<1><<<gg, NT, 0, stream>>>(vf, wvh, bv, nullptr, vtd, nullptr);

  dim3 ga(S_ / 128, H_, B_);  // (16,16,4)
  flash_attn<<<ga, NT, 0, stream>>>(qhd, khd, vtd, mask, atth);

  proj_gemm<2><<<gg, NT, 0, stream>>>(atth, woh, bo, q, nullptr, proj);

  layernorm_kernel<<<B_ * S_, NT, 0, stream>>>(proj, lng, lnb, out);
}
